// RelativeMultiHeadAttentionBlock_4286377361856
// MI455X (gfx1250) — matrix operand order repaired and checked by reading, not compiled, not run on hardware
//
// MI455X (gfx1250) Relative Multi-Head Attention (Transformer-XL style)
// B=2, T=C=1024, M=1024, S=2048, H=16, HD=64, F=1024.
//
// Key simplifications proved from the reference:
//  * K/V of mem_state are discarded by the concat/slice -> K = [cache_key, x@Wk],
//    V = [cache_value, x@Wv]; new_key/new_value = x@Wk / x@Wv.
//  * relative_shift => bias[t,s] = BD[t, s-t+1023] with BD = (q+v)@r^T; for a
//    16x32 score tile the needed BD columns form a 47-wide diagonal band that
//    fits in two rolling 32-wide WMMA-computed BD chunks (t0 is 16-aligned).
//
// ~50 GFLOP of GEMM vs ~60MB HBM traffic: compute bound -> everything runs on
// v_wmma_f32_16x16x32_bf16 (bf16 in, f32 accumulate). K/V working set per
// (b,h) = 512KB, all 32 (b,h) = 16MB << 192MB L2, so flash re-reads hit L2.

#include <hip/hip_runtime.h>
#include <stdint.h>

typedef __attribute__((ext_vector_type(16))) __bf16 v16bf;
typedef __attribute__((ext_vector_type(8)))  float  v8f;

#define BB 2
#define HH 16
#define HD 64
#define TT 1024
#define MM 1024
#define SS 2048
#define FF 1024

__device__ __forceinline__ unsigned short f2bf(float f) {
  unsigned u = __float_as_uint(f);
  unsigned r = (u + 0x7FFFu + ((u >> 16) & 1u)) >> 16;
  return (unsigned short)r;
}

union Frag16 { uint4 q[2]; v16bf v; };

__device__ __forceinline__ v8f wmma_bf16(const v16bf& a, const v16bf& b, v8f c) {
  return __builtin_amdgcn_wmma_f32_16x16x32_bf16(false, a, false, b, (short)0, c,
                                                 false, false);
}

// ---------------------------------------------------------------- prep kernels

__global__ void k_f32_to_bf16(const float* __restrict__ in,
                              unsigned short* __restrict__ out, int n) {
  int i = blockIdx.x * blockDim.x + threadIdx.x;
  if (i < n) out[i] = f2bf(in[i]);
}

// W (K=1024 x N=1024, f32, row-major) -> WT (N x K, bf16, row-major)
__global__ void k_transpose_w(const float* __restrict__ W,
                              unsigned short* __restrict__ WT) {
  __shared__ float tile[32][33];
  int bx = blockIdx.x, by = blockIdx.y;
  int tx = threadIdx.x, ty = threadIdx.y;  // (32,8)
#pragma unroll
  for (int r = 0; r < 4; r++)
    tile[ty + 8 * r][tx] = W[(size_t)(by * 32 + ty + 8 * r) * 1024 + bx * 32 + tx];
  __syncthreads();
#pragma unroll
  for (int r = 0; r < 4; r++)
    WT[(size_t)(bx * 32 + ty + 8 * r) * 1024 + by * 32 + tx] =
        f2bf(tile[tx][ty + 8 * r]);
}

// cache_key (B,C,H,HD) -> K_bf (b,h,s,hd) s<1024 ; cache_value -> VT_bf (b,h,hd,s)
__global__ void k_cache_convert(const float* __restrict__ ckey,
                                const float* __restrict__ cval,
                                unsigned short* __restrict__ Kb,
                                unsigned short* __restrict__ VT) {
  int i = blockIdx.x * blockDim.x + threadIdx.x;
  const int NK = BB * HH * TT * HD;  // 2M
  if (i < NK) {
    int hd = i & 63, t = (i >> 6) & 1023, h = (i >> 16) & 15, b = i >> 20;
    float x = ckey[(((size_t)(b * TT + t)) * HH + h) * HD + hd];
    Kb[(((size_t)(b * HH + h)) * SS + t) * HD + hd] = f2bf(x);
  } else if (i < 2 * NK) {
    int j = i - NK;
    int t = j & 1023, hd = (j >> 10) & 63, h = (j >> 16) & 15, b = j >> 20;
    float x = cval[(((size_t)(b * TT + t)) * HH + h) * HD + hd];
    VT[(((size_t)(b * HH + h)) * HD + hd) * SS + t] = f2bf(x);
  }
}

// segment ids: one block (1024 thr) per batch, Hillis-Steele scan of 3072 ints
__global__ void k_seg_scan(const int* __restrict__ mask,
                           const int* __restrict__ mem_mask,
                           const int* __restrict__ cache_mask,
                           int* __restrict__ qseg, int* __restrict__ kseg) {
  __shared__ int buf[2][3072];
  __shared__ int red[32];
  int b = blockIdx.x, tid = threadIdx.x;
  for (int i = tid; i < 3072; i += 1024) {
    int v;
    if (i < 1024)       v = mem_mask[b * 1024 + i];
    else if (i < 2048)  v = cache_mask[b * 1024 + i - 1024];
    else                v = mask[b * 1024 + i - 2048];
    buf[0][i] = v;
  }
  __syncthreads();
  int src = 0;
  for (int off = 1; off < 3072; off <<= 1) {
    for (int i = tid; i < 3072; i += 1024) {
      int v = buf[src][i];
      if (i >= off) v += buf[src][i - off];
      buf[1 - src][i] = v;
    }
    src = 1 - src;
    __syncthreads();
  }
  const int* pf = buf[src];
  int mv = -2147483647;
  for (int i = tid; i < 2048; i += 1024) mv = max(mv, pf[i]);
  for (int d = 1; d < 32; d <<= 1) mv = max(mv, __shfl_xor(mv, d, 32));
  if ((tid & 31) == 0) red[tid >> 5] = mv;
  __syncthreads();
  if (tid == 0) {
    int m2 = red[0];
    for (int w2 = 1; w2 < 32; w2++) m2 = max(m2, red[w2]);
    red[0] = m2;
  }
  __syncthreads();
  int maxv = red[0];
  int base = pf[2047];
  if (tid < 1024) qseg[b * TT + tid] = pf[2048 + tid] - base + maxv;
  for (int i = tid; i < 2048; i += 1024) kseg[b * SS + i] = pf[1024 + i];
}

// ------------------------------------------------- generic bf16 WMMA GEMM
// C(2048x1024) = A(2048x1024,bf16) @ WT^T ; WT stored (N x K) bf16.
// Per-wave 64x64 tile = 4x4 fragments, K-loop of 32 (32 iters, 16 WMMA each).
#define GM_R 0
#define GM_Q 1
#define GM_K 2
#define GM_V 3
#define GM_O 4

__global__ __launch_bounds__(128) void k_gemm_bf16(
    const unsigned short* __restrict__ A, const unsigned short* __restrict__ WT,
    int mode, unsigned short* __restrict__ outA, unsigned short* __restrict__ outB,
    float* __restrict__ outF, const float* __restrict__ uvec,
    const float* __restrict__ vvec) {
  const int Kdim = 1024;
  int wave = threadIdx.x >> 5, lane = threadIdx.x & 31;
  int l16 = lane & 15, hiA = lane >> 4;
  int tile = blockIdx.x * 4 + wave;  // 512 tiles: 32(M) x 16(N)
  int m0 = (tile >> 4) * 64, n0 = (tile & 15) * 64;

  v8f c[4][4];
#pragma unroll
  for (int i = 0; i < 4; i++)
#pragma unroll
    for (int j = 0; j < 4; j++) {
      v8f z = {0.f, 0.f, 0.f, 0.f, 0.f, 0.f, 0.f, 0.f};
      c[i][j] = z;
    }

  for (int k0 = 0; k0 < Kdim; k0 += 32) {
    Frag16 a[4], bf[4];
#pragma unroll
    for (int i = 0; i < 4; i++) {
      const unsigned short* p =
          A + (size_t)(m0 + 16 * i + l16) * Kdim + k0 + (hiA ? 16 : 0);
      a[i].q[0] = *(const uint4*)p;
      a[i].q[1] = *(const uint4*)(p + 8);
    }
#pragma unroll
    for (int j = 0; j < 4; j++) {
      const unsigned short* p =
          WT + (size_t)(n0 + 16 * j + l16) * Kdim + k0 + (hiA ? 16 : 0);
      bf[j].q[0] = *(const uint4*)p;
      bf[j].q[1] = *(const uint4*)(p + 8);
    }
#pragma unroll
    for (int i = 0; i < 4; i++)
#pragma unroll
      for (int j = 0; j < 4; j++) c[i][j] = wmma_bf16(a[i].v, bf[j].v, c[i][j]);
  }

  int rbase = hiA ? 8 : 0;
#pragma unroll
  for (int i = 0; i < 4; i++)
#pragma unroll
    for (int j = 0; j < 4; j++) {
      int col = n0 + 16 * j + l16;
#pragma unroll
      for (int e = 0; e < 8; e++) {
        int row = m0 + 16 * i + rbase + e;
        float val = c[i][j][e];
        int h = col >> 6, hd = col & 63, b = row >> 10, t = row & 1023;
        if (mode == GM_R) {
          outA[((size_t)h * SS + row) * HD + hd] = f2bf(val);
        } else if (mode == GM_Q) {
          size_t o = (((size_t)(b * HH + h)) * TT + t) * HD + hd;
          outA[o] = f2bf((val + uvec[col]) * 0.125f);   // (q+u)*scale
          outB[o] = f2bf((val + vvec[col]) * 0.125f);   // (q+v)*scale
        } else if (mode == GM_K) {
          outA[(((size_t)(b * HH + h)) * SS + MM + t) * HD + hd] = f2bf(val);
          outF[(((size_t)(b * TT + t)) * HH + h) * HD + hd] = val;  // new_key
        } else if (mode == GM_V) {
          outA[(((size_t)(b * HH + h)) * HD + hd) * SS + MM + t] = f2bf(val);
          outF[(((size_t)(b * TT + t)) * HH + h) * HD + hd] = val;  // new_value
        } else {  // GM_O
          outF[(size_t)row * FF + col] = val;
        }
      }
    }
}

// ----------------------------------------------------------- flash attention
// 4 waves/block; wave owns 16 query rows (t0), loops s-chunks of 32.
// Per chunk: 4 WMMA (rolling BD bias tile) + 4 WMMA (AC=QuK^T) + 4 WMMA (PV).
__global__ __launch_bounds__(128) void k_flash(
    const unsigned short* __restrict__ Qu, const unsigned short* __restrict__ Qv,
    const unsigned short* __restrict__ Kb, const unsigned short* __restrict__ VT,
    const unsigned short* __restrict__ Rb, const int* __restrict__ qseg,
    const int* __restrict__ kseg, unsigned short* __restrict__ Obf) {
  __shared__ __align__(16) float bdLds[4][2][16 * 32];
  __shared__ __align__(16) unsigned short pLds[4][16 * 32];

  int wave = threadIdx.x >> 5, lane = threadIdx.x & 31;
  int l16 = lane & 15, hiA = lane >> 4;
  int bh = blockIdx.x >> 4, tblk = blockIdx.x & 15;
  int b = bh >> 4, h = bh & 15;
  int t0 = tblk * 64 + wave * 16;
  int rowBase = hiA ? 8 : 0;

  float* bd0 = bdLds[wave][0];
  float* bd1 = bdLds[wave][1];
  unsigned short* pT = pLds[wave];

  const unsigned short* QuBase = Qu + (((size_t)(b * HH + h)) * TT + t0 + l16) * HD;
  const unsigned short* QvBase = Qv + (((size_t)(b * HH + h)) * TT + t0 + l16) * HD;
  Frag16 aU[2], aV[2];
#pragma unroll
  for (int kh = 0; kh < 2; kh++) {
    const unsigned short* pu = QuBase + kh * 32 + (hiA ? 16 : 0);
    aU[kh].q[0] = *(const uint4*)pu;
    aU[kh].q[1] = *(const uint4*)(pu + 8);
    const unsigned short* pv = QvBase + kh * 32 + (hiA ? 16 : 0);
    aV[kh].q[0] = *(const uint4*)pv;
    aV[kh].q[1] = *(const uint4*)(pv + 8);
  }

  int qsg[8];
#pragma unroll
  for (int e = 0; e < 8; e++) qsg[e] = qseg[b * TT + t0 + rowBase + e];

  float mrow[8], lrow[8];
#pragma unroll
  for (int e = 0; e < 8; e++) { mrow[e] = -3.0e38f; lrow[e] = 0.f; }
  v8f o[4];
#pragma unroll
  for (int j = 0; j < 4; j++) {
    v8f z = {0.f, 0.f, 0.f, 0.f, 0.f, 0.f, 0.f, 0.f};
    o[j] = z;
  }

  int a_off = (1008 - t0) & 31;          // in {0,16} -> 47-wide band fits 64
  int moff = (1008 - t0) - a_off;

  const unsigned short* Rbase = Rb + (size_t)h * SS * HD;
  const unsigned short* Kbase = Kb + ((size_t)(b * HH + h)) * SS * HD;
  const unsigned short* Vbase = VT + ((size_t)(b * HH + h)) * HD * SS;
  const int* ksegb = kseg + b * SS;

  auto computeBD = [&](int j, float* slot) {
    int mb = moff + 32 * j;
#pragma unroll
    for (int nh = 0; nh < 2; nh++) {
      v8f cb = {0.f, 0.f, 0.f, 0.f, 0.f, 0.f, 0.f, 0.f};
      int m = mb + nh * 16 + l16;
      m = min(max(m, 0), SS - 1);  // OOB rows are masked later anyway
#pragma unroll
      for (int kh = 0; kh < 2; kh++) {
        const unsigned short* p = Rbase + (size_t)m * HD + kh * 32 + (hiA ? 16 : 0);
        Frag16 bbf;
        bbf.q[0] = *(const uint4*)p;
        bbf.q[1] = *(const uint4*)(p + 8);
        cb = wmma_bf16(aV[kh].v, bbf.v, cb);
      }
#pragma unroll
      for (int e = 0; e < 8; e++)
        slot[(rowBase + e) * 32 + nh * 16 + l16] = cb[e];
    }
  };

  int c_last = (t0 + 15 + 1024) >> 5;
  if (c_last > 63) c_last = 63;

  computeBD(0, bd0);
  computeBD(1, bd1);

  for (int cc = 0; cc <= c_last; ++cc) {
    float* lo = (cc & 1) ? bd1 : bd0;   // chunk cc
    float* hi = (cc & 1) ? bd0 : bd1;   // chunk cc+1
    asm volatile("s_wait_dscnt 0x0" ::: "memory");
    int s0 = cc * 32;

    float lg[2][8];
#pragma unroll
    for (int nh = 0; nh < 2; nh++) {
      v8f ca = {0.f, 0.f, 0.f, 0.f, 0.f, 0.f, 0.f, 0.f};
#pragma unroll
      for (int kh = 0; kh < 2; kh++) {
        const unsigned short* p =
            Kbase + (size_t)(s0 + nh * 16 + l16) * HD + kh * 32 + (hiA ? 16 : 0);
        Frag16 bbf;
        bbf.q[0] = *(const uint4*)p;
        bbf.q[1] = *(const uint4*)(p + 8);
        ca = wmma_bf16(aU[kh].v, bbf.v, ca);
      }
      int ksg = ksegb[s0 + nh * 16 + l16];
      int sl = nh * 16 + l16;
#pragma unroll
      for (int e = 0; e < 8; e++) {
        int tl = rowBase + e;
        int rel = sl - tl + 15 + a_off;  // in [a, a+46] subset of [0,63]
        float bias = (rel < 32) ? lo[tl * 32 + rel] : hi[tl * 32 + rel - 32];
        bool keep = (s0 + sl <= t0 + tl + 1024) && (ksg == qsg[e]);
        lg[nh][e] = keep ? (ca[e] + bias) : -1.0e30f;
      }
    }

    float al[8];
#pragma unroll
    for (int e = 0; e < 8; e++) {
      float mx = fmaxf(lg[0][e], lg[1][e]);
      mx = fmaxf(mx, __shfl_xor(mx, 1, 32));
      mx = fmaxf(mx, __shfl_xor(mx, 2, 32));
      mx = fmaxf(mx, __shfl_xor(mx, 4, 32));
      mx = fmaxf(mx, __shfl_xor(mx, 8, 32));
      float mnew = fmaxf(mrow[e], mx);
      float alpha = __expf(mrow[e] - mnew);
      float p0 = (lg[0][e] < -1.0e29f) ? 0.f : __expf(lg[0][e] - mnew);
      float p1 = (lg[1][e] < -1.0e29f) ? 0.f : __expf(lg[1][e] - mnew);
      float sm = p0 + p1;
      sm += __shfl_xor(sm, 1, 32);
      sm += __shfl_xor(sm, 2, 32);
      sm += __shfl_xor(sm, 4, 32);
      sm += __shfl_xor(sm, 8, 32);
      lrow[e] = lrow[e] * alpha + sm;
      mrow[e] = mnew;
      al[e] = alpha;
      int tl = rowBase + e;
      pT[tl * 32 + l16] = f2bf(p0);
      pT[tl * 32 + 16 + l16] = f2bf(p1);
    }

#pragma unroll
    for (int j = 0; j < 4; j++)
#pragma unroll
      for (int e = 0; e < 8; e++) o[j][e] *= al[e];

    asm volatile("s_wait_dscnt 0x0" ::: "memory");
    Frag16 ap;
    {
      const unsigned short* pp = pT + l16 * 32 + (hiA ? 16 : 0);
      ap.q[0] = *(const uint4*)pp;
      ap.q[1] = *(const uint4*)(pp + 8);
    }
#pragma unroll
    for (int j = 0; j < 4; j++) {
      const unsigned short* p =
          Vbase + (size_t)(j * 16 + l16) * SS + s0 + (hiA ? 16 : 0);
      Frag16 bbf;
      bbf.q[0] = *(const uint4*)p;
      bbf.q[1] = *(const uint4*)(p + 8);
      o[j] = wmma_bf16(ap.v, bbf.v, o[j]);
    }

    if (cc + 2 <= c_last + 1) computeBD(cc + 2, lo);  // same parity slot
  }

#pragma unroll
  for (int e = 0; e < 8; e++) lrow[e] = 1.0f / lrow[e];
#pragma unroll
  for (int j = 0; j < 4; j++)
#pragma unroll
    for (int e = 0; e < 8; e++) {
      int t = t0 + rowBase + e;
      int hd = j * 16 + l16;
      Obf[((size_t)(b * TT + t)) * FF + h * HD + hd] = f2bf(o[j][e] * lrow[e]);
    }
}

// ------------------------------------------------------------------- launch

extern "C" void kernel_launch(void* const* d_in, const int* in_sizes, int n_in,
                              void* d_out, int out_size, void* d_ws,
                              size_t ws_size, hipStream_t stream) {
  (void)in_sizes; (void)n_in; (void)out_size; (void)ws_size;
  const float* x          = (const float*)d_in[0];
  const int*   mask       = (const int*)d_in[1];
  /* d_in[2] mem_state: its K/V are discarded by the reference slicing */
  const int*   mem_mask   = (const int*)d_in[3];
  const float* cache_key  = (const float*)d_in[4];
  const float* cache_val  = (const float*)d_in[5];
  const int*   cache_mask = (const int*)d_in[6];
  const float* rel_emb    = (const float*)d_in[7];
  const float* Wq = (const float*)d_in[8];
  const float* Wk = (const float*)d_in[9];
  const float* Wv = (const float*)d_in[10];
  const float* Wr = (const float*)d_in[11];
  const float* u  = (const float*)d_in[12];
  const float* v  = (const float*)d_in[13];
  const float* Wo = (const float*)d_in[14];

  float* out_f   = (float*)d_out;                  // (B,T,F)
  float* new_key = out_f + (size_t)BB * TT * FF;   // (B,C,H,HD)
  float* new_val = new_key + (size_t)BB * TT * HH * HD;

  char* w = (char*)d_ws;
  auto alloc = [&](size_t bytes) -> char* {
    char* p = w;
    w += (bytes + 255) & ~(size_t)255;
    return p;
  };
  const size_t NE = (size_t)2048 * 1024;  // rows x K elements
  unsigned short* x_bf  = (unsigned short*)alloc(NE * 2);
  unsigned short* re_bf = (unsigned short*)alloc(NE * 2);
  unsigned short* WqT = (unsigned short*)alloc(1024 * 1024 * 2);
  unsigned short* WkT = (unsigned short*)alloc(1024 * 1024 * 2);
  unsigned short* WvT = (unsigned short*)alloc(1024 * 1024 * 2);
  unsigned short* WrT = (unsigned short*)alloc(1024 * 1024 * 2);
  unsigned short* WoT = (unsigned short*)alloc(1024 * 1024 * 2);
  unsigned short* Qu_bf = (unsigned short*)alloc(NE * 2);
  unsigned short* Qv_bf = (unsigned short*)alloc(NE * 2);
  unsigned short* K_bf  = (unsigned short*)alloc((size_t)BB * HH * SS * HD * 2);
  unsigned short* VT_bf = (unsigned short*)alloc((size_t)BB * HH * SS * HD * 2);
  unsigned short* R_bf  = (unsigned short*)alloc((size_t)HH * SS * HD * 2);
  unsigned short* O_bf  = (unsigned short*)alloc(NE * 2);
  int* qseg = (int*)alloc((size_t)BB * TT * 4);
  int* kseg = (int*)alloc((size_t)BB * SS * 4);

  const int n2m = (int)NE;  // 2,097,152
  hipLaunchKernelGGL(k_f32_to_bf16, dim3((n2m + 255) / 256), dim3(256), 0, stream,
                     x, x_bf, n2m);
  hipLaunchKernelGGL(k_f32_to_bf16, dim3((n2m + 255) / 256), dim3(256), 0, stream,
                     rel_emb, re_bf, n2m);
  hipLaunchKernelGGL(k_transpose_w, dim3(32, 32), dim3(32, 8), 0, stream, Wq, WqT);
  hipLaunchKernelGGL(k_transpose_w, dim3(32, 32), dim3(32, 8), 0, stream, Wk, WkT);
  hipLaunchKernelGGL(k_transpose_w, dim3(32, 32), dim3(32, 8), 0, stream, Wv, WvT);
  hipLaunchKernelGGL(k_transpose_w, dim3(32, 32), dim3(32, 8), 0, stream, Wr, WrT);
  hipLaunchKernelGGL(k_transpose_w, dim3(32, 32), dim3(32, 8), 0, stream, Wo, WoT);
  hipLaunchKernelGGL(k_seg_scan, dim3(BB), dim3(1024), 0, stream, mask, mem_mask,
                     cache_mask, qseg, kseg);
  hipLaunchKernelGGL(k_cache_convert, dim3((2 * n2m + 255) / 256), dim3(256), 0,
                     stream, cache_key, cache_val, K_bf, VT_bf);

  hipLaunchKernelGGL(k_gemm_bf16, dim3(128), dim3(128), 0, stream, x_bf, WqT,
                     GM_Q, Qu_bf, Qv_bf, (float*)nullptr, u, v);
  hipLaunchKernelGGL(k_gemm_bf16, dim3(128), dim3(128), 0, stream, x_bf, WkT,
                     GM_K, K_bf, (unsigned short*)nullptr, new_key,
                     (const float*)nullptr, (const float*)nullptr);
  hipLaunchKernelGGL(k_gemm_bf16, dim3(128), dim3(128), 0, stream, x_bf, WvT,
                     GM_V, VT_bf, (unsigned short*)nullptr, new_val,
                     (const float*)nullptr, (const float*)nullptr);
  hipLaunchKernelGGL(k_gemm_bf16, dim3(128), dim3(128), 0, stream, re_bf, WrT,
                     GM_R, R_bf, (unsigned short*)nullptr, (float*)nullptr,
                     (const float*)nullptr, (const float*)nullptr);

  hipLaunchKernelGGL(k_flash, dim3(512), dim3(128), 0, stream, Qu_bf, Qv_bf,
                     K_bf, VT_bf, R_bf, qseg, kseg, O_bf);

  hipLaunchKernelGGL(k_gemm_bf16, dim3(128), dim3(128), 0, stream, O_bf, WoT,
                     GM_O, (unsigned short*)nullptr, (unsigned short*)nullptr,
                     out_f, (const float*)nullptr, (const float*)nullptr);
}
